// Cos_Attn_22393959481496
// MI455X (gfx1250) — matrix operand order repaired and checked by reading, not compiled, not run on hardware
//
#include <hip/hip_runtime.h>

typedef __attribute__((ext_vector_type(16))) _Float16 v16h;
typedef __attribute__((ext_vector_type(8)))  _Float16 v8h;
typedef __attribute__((ext_vector_type(8)))  float    v8f;

#define N_SP  4096   // W*H
#define C_DIM 512    // channels (K dimension)

// ---------------------------------------------------------------------------
// Kernel 1: compute per-spatial-position L2 norms of x[:, n], and write the
// normalized, transposed matrix qh[n][c] = x[c][n] / ||x[:,n]|| as f16,
// row-major [N, C] so WMMA fragments load as contiguous b128s.
// ---------------------------------------------------------------------------
__global__ void __launch_bounds__(256) normalize_qh(const float* __restrict__ x,
                                                    _Float16* __restrict__ qh) {
  __shared__ float red[8][32];
  __shared__ float rnorm[32];
  __shared__ float tile[32][33];   // +1 pad to avoid LDS bank conflicts

  const int tid = threadIdx.x;
  const int tx  = tid & 31;        // spatial offset within block's 32 columns
  const int ty  = tid >> 5;        // 0..7
  const int n0  = blockIdx.x * 32;

  // Pass A: sum of squares down the channel dim (coalesced along n)
  float ss = 0.f;
  for (int c = ty; c < C_DIM; c += 8) {
    float v = x[(size_t)c * N_SP + n0 + tx];
    ss += v * v;
  }
  red[ty][tx] = ss;
  __syncthreads();
  if (ty == 0) {
    float t = 0.f;
#pragma unroll
    for (int j = 0; j < 8; ++j) t += red[j][tx];
    rnorm[tx] = rsqrtf(t);
  }
  __syncthreads();

  // Pass B: 32x32 tiled transpose, scale by 1/norm, convert to f16
  for (int c0 = 0; c0 < C_DIM; c0 += 32) {
#pragma unroll
    for (int i = 0; i < 4; ++i) {
      tile[ty + 8 * i][tx] = x[(size_t)(c0 + ty + 8 * i) * N_SP + n0 + tx];
    }
    __syncthreads();
#pragma unroll
    for (int i = 0; i < 4; ++i) {
      const int nn = ty + 8 * i;   // spatial index within tile
      qh[(size_t)(n0 + nn) * C_DIM + c0 + tx] =
          (_Float16)(tile[tx][nn] * rnorm[nn]);
    }
    __syncthreads();
  }
}

// ---------------------------------------------------------------------------
// WMMA fragment loaders for V_WMMA_F32_16X16X32_F16 (ISA 7.12.2 layouts).
// A (16x32, MxK): lane<16 holds row M=lane, K = {kb+0..7, kb+16..23}
//                 lane>=16 holds row M=lane-16, K = {kb+8..15, kb+24..31}
// B (32x16, KxN): lane<16 holds col N=lane, K = {kb+0..7, kb+16..23}
//                 lane>=16 holds col N=lane-16, K = {kb+8..15, kb+24..31}
// Since S = Qh * Qh^T, B's column n is Qh row (tn + n): both fragments load
// from the same row-major qh with aligned 16-byte chunks.
// ---------------------------------------------------------------------------
__device__ __forceinline__ v16h load_frag_a(const _Float16* __restrict__ qh,
                                            int rowBase, int kb, int lane) {
  const int r    = rowBase + (lane & 15);
  const int koff = (lane < 16) ? 0 : 8;
  const _Float16* p = qh + (size_t)r * C_DIM + kb + koff;
  union { v16h v; v8h h[2]; } u;
  u.h[0] = *(const v8h*)(p);        // K = koff .. koff+7
  u.h[1] = *(const v8h*)(p + 16);   // K = koff+16 .. koff+23
  return u.v;
}

__device__ __forceinline__ v16h load_frag_b(const _Float16* __restrict__ qh,
                                            int rowBase, int kb, int lane) {
  const int r    = rowBase + (lane & 15);
  const int koff = (lane < 16) ? 0 : 8;
  const _Float16* p = qh + (size_t)r * C_DIM + kb + koff;
  union { v16h v; v8h h[2]; } u;
  u.h[0] = *(const v8h*)(p);        // K = koff .. koff+7
  u.h[1] = *(const v8h*)(p + 16);   // K = koff+16 .. koff+23
  return u.v;
}

// ---------------------------------------------------------------------------
// Kernel 2: Gram matrix S = Qh * Qh^T via v_wmma_f32_16x16x32_f16.
// Each wave computes a 64x64 output tile (4x4 fragment block, 16 f32 accums:
// 128 accum VGPRs + 64 fragment VGPRs, no spill). Per K-step: 8 fragment
// loads (16 x b128) feed 16 WMMAs -> 2x the FLOP/byte of a 32x32 tile,
// halving L2 fragment traffic for this L2-resident GEMM.
// ---------------------------------------------------------------------------
__global__ void __launch_bounds__(256) gram_wmma(const _Float16* __restrict__ qh,
                                                 float* __restrict__ out) {
  const int lane = threadIdx.x & 31;
  const int wid  = blockIdx.x * (blockDim.x >> 5) + (threadIdx.x >> 5);
  const int tilesPerRow = N_SP / 64;            // 64
  const int tm = (wid / tilesPerRow) * 64;
  const int tn = (wid % tilesPerRow) * 64;

  v8f acc[4][4];
#pragma unroll
  for (int i = 0; i < 4; ++i)
#pragma unroll
    for (int j = 0; j < 4; ++j) acc[i][j] = (v8f){};

  for (int kb = 0; kb < C_DIM; kb += 32) {
    v16h a[4], b[4];
#pragma unroll
    for (int i = 0; i < 4; ++i) a[i] = load_frag_a(qh, tm + 16 * i, kb, lane);
#pragma unroll
    for (int j = 0; j < 4; ++j) b[j] = load_frag_b(qh, tn + 16 * j, kb, lane);
#pragma unroll
    for (int i = 0; i < 4; ++i)
#pragma unroll
      for (int j = 0; j < 4; ++j)
        acc[i][j] = __builtin_amdgcn_wmma_f32_16x16x32_f16(
            false, a[i], false, b[j], (short)0, acc[i][j], false, false);
  }

  // C/D layout: VGPR r, lanes 0-15 -> row r, lanes 16-31 -> row r+8; col = lane&15
  const int rofs = (lane >> 4) << 3;   // 0 or 8
  const int col  = lane & 15;
#pragma unroll
  for (int i = 0; i < 4; ++i)
#pragma unroll
    for (int j = 0; j < 4; ++j)
#pragma unroll
      for (int r = 0; r < 8; ++r)
        out[(size_t)(tm + 16 * i + r + rofs) * N_SP + tn + 16 * j + col] =
            acc[i][j][r];
}

// ---------------------------------------------------------------------------
// Kernel 3: in-place row softmax over d_out (4096 rows of 4096 f32).
// One 256-thread block per row; 16 values per thread kept in registers.
// ---------------------------------------------------------------------------
__global__ void __launch_bounds__(256) row_softmax(float* __restrict__ out) {
  __shared__ float sdata[256];
  const int tid = threadIdx.x;
  float* rp = out + (size_t)blockIdx.x * N_SP;

  float vals[16];
  float m = -3.402823466e38f;
#pragma unroll
  for (int j = 0; j < 16; ++j) {
    vals[j] = rp[tid + 256 * j];
    m = fmaxf(m, vals[j]);
  }
  sdata[tid] = m;
  __syncthreads();
  for (int s = 128; s > 0; s >>= 1) {
    if (tid < s) sdata[tid] = fmaxf(sdata[tid], sdata[tid + s]);
    __syncthreads();
  }
  m = sdata[0];
  __syncthreads();

  float sum = 0.f;
#pragma unroll
  for (int j = 0; j < 16; ++j) {
    vals[j] = __expf(vals[j] - m);
    sum += vals[j];
  }
  sdata[tid] = sum;
  __syncthreads();
  for (int s = 128; s > 0; s >>= 1) {
    if (tid < s) sdata[tid] += sdata[tid + s];
    __syncthreads();
  }
  const float inv = 1.0f / sdata[0];
#pragma unroll
  for (int j = 0; j < 16; ++j) rp[tid + 256 * j] = vals[j] * inv;
}

// ---------------------------------------------------------------------------
extern "C" void kernel_launch(void* const* d_in, const int* in_sizes, int n_in,
                              void* d_out, int out_size, void* d_ws, size_t ws_size,
                              hipStream_t stream) {
  const float* x   = (const float*)d_in[0];
  float*       out = (float*)d_out;
  _Float16*    qh  = (_Float16*)d_ws;   // needs N_SP*C_DIM*2 = 4 MB scratch

  (void)in_sizes; (void)n_in; (void)out_size; (void)ws_size;

  // 1) norms + normalized f16 transpose: 128 blocks x 32 spatial columns
  normalize_qh<<<N_SP / 32, 256, 0, stream>>>(x, qh);

  // 2) Gram matrix: (4096/64)^2 = 4096 waves, 8 waves per 256-thread block
  gram_wmma<<<(N_SP / 64) * (N_SP / 64) / 8, 256, 0, stream>>>(qh, out);

  // 3) row-wise softmax in place
  row_softmax<<<N_SP, 256, 0, stream>>>(out);
}
